// DoubleTrans_171798692595
// MI455X (gfx1250) — matrix operand order repaired and checked by reading, not compiled, not run on hardware
//
#include <hip/hip_runtime.h>
#include <hip/hip_bf16.h>
#include <utility>

typedef __attribute__((ext_vector_type(16))) __bf16 v16bf;
typedef __attribute__((ext_vector_type(8)))  float  v8f;

#define BATCH 8
#define NSTATE 32

__device__ __forceinline__ float gelu_f(float x) {
    const float c = 0.7978845608028654f;
    float t = tanhf(c * (x + 0.044715f * x * x * x));
    return 0.5f * x * (1.0f + t);
}

// ---------------------------------------------------------------- transpose in
__global__ void k_transpose_in(const float* __restrict__ x, float* __restrict__ u,
                               int L, int D) {
    size_t i = (size_t)blockIdx.x * blockDim.x + threadIdx.x;
    size_t n = (size_t)BATCH * L * D;
    if (i >= n) return;
    size_t b = i / ((size_t)L * D);
    size_t r = i % ((size_t)L * D);
    size_t l = r / D, d = r % D;
    u[(b * D + d) * L + l] = x[i];
}

// ---------------------------------------------------------------- copy / add
__global__ void k_copy(float* __restrict__ dst, const float* __restrict__ src, size_t n) {
    size_t i = (size_t)blockIdx.x * blockDim.x + threadIdx.x;
    if (i < n) dst[i] = src[i];
}
__global__ void k_add(float* __restrict__ out, const float* __restrict__ a,
                      const float* __restrict__ b, size_t n) {
    size_t i = (size_t)blockIdx.x * blockDim.x + threadIdx.x;
    if (i < n) out[i] = a[i] + b[i];
}

// ---------------------------------------------------------------- channel LN
__global__ void k_ln_chan(const float* __restrict__ x, float* __restrict__ z,
                          const float* __restrict__ g, const float* __restrict__ b,
                          int H, int L) {
    int i = blockIdx.x * blockDim.x + threadIdx.x;
    if (i >= BATCH * L) return;
    int bb = i / L, l = i % L;
    size_t base = (size_t)bb * H * L + l;
    float m = 0.f;
    for (int h = 0; h < H; ++h) m += x[base + (size_t)h * L];
    m /= (float)H;
    float v = 0.f;
    for (int h = 0; h < H; ++h) { float d = x[base + (size_t)h * L] - m; v += d * d; }
    v /= (float)H;
    float rs = rsqrtf(v + 1e-5f);
    for (int h = 0; h < H; ++h)
        z[base + (size_t)h * L] = (x[base + (size_t)h * L] - m) * rs * g[h] + b[h];
}

// ---------------------------------------------------------------- S4 Ct + dtA
// CtBuf[(h*32+n)*4] = {Ct_re, Ct_im, dt*A_re, dt*A_im}
__global__ void k_ct_gen(const float* __restrict__ log_dt,
                         const float* __restrict__ A_re, const float* __restrict__ A_im,
                         const float* __restrict__ B_re, const float* __restrict__ B_im,
                         const float* __restrict__ C_re, const float* __restrict__ C_im,
                         float* __restrict__ ct, int H) {
    int i = blockIdx.x * blockDim.x + threadIdx.x;
    if (i >= H * NSTATE) return;
    int h = i / NSTATE;
    float dt  = expf(log_dt[h]);
    float ar  = -expf(A_re[i]);
    float ai  = A_im[i];
    float dar = dt * ar, dai = dt * ai;
    float ex  = expf(dar);
    float er  = ex * cosf(dai) - 1.0f;   // exp(dtA)-1
    float ei  = ex * sinf(dai);
    float br = B_re[i], bi = B_im[i], cr = C_re[i], ci = C_im[i];
    float pr = cr * br - ci * bi;        // C*B
    float pi = cr * bi + ci * br;
    float nr = pr * er - pi * ei;        // C*B*(exp(dtA)-1)
    float ni = pr * ei + pi * er;
    float den = ar * ar + ai * ai;
    float ctr = (nr * ar + ni * ai) / den;   // divide by A
    float cti = (ni * ar - nr * ai) / den;
    ct[i * 4 + 0] = ctr; ct[i * 4 + 1] = cti;
    ct[i * 4 + 2] = dar; ct[i * 4 + 3] = dai;
}

// K[h,l] = 2 * Re( sum_n Ct[h,n] * exp(dtA[h,n] * l) )
__global__ void k_kgen(const float* __restrict__ ct, float* __restrict__ Kf,
                       int H, int L) {
    size_t i = (size_t)blockIdx.x * blockDim.x + threadIdx.x;
    if (i >= (size_t)H * L) return;
    int h = (int)(i / L);
    float fl = (float)(i % L);
    float acc = 0.f;
    const float* p = ct + (size_t)h * NSTATE * 4;
#pragma unroll 4
    for (int n = 0; n < NSTATE; ++n) {
        float ctr = p[n * 4 + 0], cti = p[n * 4 + 1];
        float dar = p[n * 4 + 2], dai = p[n * 4 + 3];
        float e = expf(dar * fl);
        float s = sinf(dai * fl), c = cosf(dai * fl);
        acc += e * (ctr * c - cti * s);
    }
    Kf[i] = 2.0f * acc;
}

// ---------------------------------------------------------------- causal conv
// y[b,h,l] = sum_{j=0..l} K[h,j] * z[b,h,l-j].  L multiple of 256.
__global__ void k_conv(const float* __restrict__ Z, const float* __restrict__ Kf,
                       float* __restrict__ Y, int H, int L) {
    __shared__ float zs[256];
    __shared__ float ks[511];
    int bh = blockIdx.y;
    int b = bh / H, h = bh % H;
    int t0 = blockIdx.x * 256;
    int lt = threadIdx.x;
    size_t zbase = ((size_t)b * H + h) * L;
    size_t kbase = (size_t)h * L;
    float acc = 0.f;
    for (int s0 = 0; s0 <= t0; s0 += 256) {
        int d = t0 - s0;
        __syncthreads();
        zs[lt] = Z[zbase + s0 + lt];
        if (s0 + 256 <= t0) __builtin_prefetch(&Z[zbase + s0 + 256 + lt], 0, 1);
        for (int i = lt; i < 511; i += 256) {
            int idx = d - 255 + i;
            ks[i] = (idx >= 0 && idx < L) ? Kf[kbase + idx] : 0.f;
        }
        __syncthreads();
#pragma unroll 8
        for (int j = 0; j < 256; ++j) acc = fmaf(zs[j], ks[lt + 255 - j], acc);
    }
    Y[zbase + t0 + lt] = acc;
}

// y = gelu(y + D[h]*z)
__global__ void k_dskip_gelu(float* __restrict__ y, const float* __restrict__ z,
                             const float* __restrict__ D, int H, int L, size_t n) {
    size_t i = (size_t)blockIdx.x * blockDim.x + threadIdx.x;
    if (i >= n) return;
    int h = (int)((i / L) % H);
    y[i] = gelu_f(y[i] + D[h] * z[i]);
}

// out = x + a * sigmoid(g); yw is (B, 2H, L)
__global__ void k_glu(float* __restrict__ out, const float* __restrict__ x,
                      const float* __restrict__ yw, int H, int L, size_t n) {
    size_t i = (size_t)blockIdx.x * blockDim.x + threadIdx.x;
    if (i >= n) return;
    size_t hl = i % ((size_t)H * L);
    size_t b  = i / ((size_t)H * L);
    size_t wb = b * 2 * (size_t)H * L;
    float a = yw[wb + hl];
    float g = yw[wb + (size_t)H * L + hl];
    out[i] = x[i] + a * (1.0f / (1.0f + expf(-g)));
}

// ---------------------------------------------------------------- WMMA GEMM
// out[b,o,l] = act( sum_h W[o,h]*X[b,h,l] + bias[o] ) (+ res[b,o,l])
// Block: 256 threads (8 waves) computes 128(O) x 16(L) tile for one batch.
// A fragments: direct vectorized global b128 loads (L2-hot weight panel).
// B fragments: X panel staged transposed in LDS -> ds_load_b128 per lane.
// O mult of 128, Kd mult of 32, L mult of 16.
__global__ void __launch_bounds__(256)
k_gemm_wmma(const float* __restrict__ W, const float* __restrict__ X,
            const float* __restrict__ bias, const float* __restrict__ res,
            float* __restrict__ out, int O, int Kd, int L, int act) {
    __shared__ float Xt[16 * 36];    // transposed X panel: 16 n-rows x 32 k, stride 36

    int tilesL = L / 16;
    int to = blockIdx.x / tilesL;    // 128-row O block
    int tl = blockIdx.x % tilesL;    // 16-col L tile
    int b  = blockIdx.y;
    int tid  = threadIdx.x;
    int wid  = tid >> 5;
    int lane = tid & 31;
    int row  = lane & 15;
    int hi   = lane >> 4;

    const float* Wrow  = W + (size_t)(to * 128 + wid * 16 + row) * Kd + hi * 8;
    const float* Xbase = X + (size_t)b * Kd * L + tl * 16;

    v8f acc = {};
    for (int k0 = 0; k0 < Kd; k0 += 32) {
        __syncthreads();
        // stage X panel 32x16 transposed: one b128 global load per thread (tid<128)
        if (tid < 128) {
            int kk = tid >> 2, c = (tid & 3) * 4;   // kk 0..31, c 0,4,8,12
            float4 xv = *(const float4*)(Xbase + (size_t)(k0 + kk) * L + c);
            Xt[(c + 0) * 36 + kk] = xv.x;
            Xt[(c + 1) * 36 + kk] = xv.y;
            Xt[(c + 2) * 36 + kk] = xv.z;
            Xt[(c + 3) * 36 + kk] = xv.w;
        }
        if (k0 + 32 < Kd) __builtin_prefetch(Wrow + k0 + 32, 0, 1);
        __syncthreads();

        // A fragment: two 8-float runs, 16B-aligned vector loads from global
        float4 a0 = *(const float4*)(Wrow + k0);
        float4 a1 = *(const float4*)(Wrow + k0 + 4);
        float4 a2 = *(const float4*)(Wrow + k0 + 16);
        float4 a3 = *(const float4*)(Wrow + k0 + 20);
        // B fragment: two 8-k runs (8*hi, 16+8*hi) at row `row` -> 4x ds_load_b128
        const float4* xp = (const float4*)&Xt[row * 36 + hi * 8];
        float4 b0 = xp[0], b1 = xp[1], b2 = xp[4], b3 = xp[5];

        v16bf afrag, bfrag;
        afrag[0]  = (__bf16)a0.x; afrag[1]  = (__bf16)a0.y;
        afrag[2]  = (__bf16)a0.z; afrag[3]  = (__bf16)a0.w;
        afrag[4]  = (__bf16)a1.x; afrag[5]  = (__bf16)a1.y;
        afrag[6]  = (__bf16)a1.z; afrag[7]  = (__bf16)a1.w;
        afrag[8]  = (__bf16)a2.x; afrag[9]  = (__bf16)a2.y;
        afrag[10] = (__bf16)a2.z; afrag[11] = (__bf16)a2.w;
        afrag[12] = (__bf16)a3.x; afrag[13] = (__bf16)a3.y;
        afrag[14] = (__bf16)a3.z; afrag[15] = (__bf16)a3.w;
        bfrag[0]  = (__bf16)b0.x; bfrag[1]  = (__bf16)b0.y;
        bfrag[2]  = (__bf16)b0.z; bfrag[3]  = (__bf16)b0.w;
        bfrag[4]  = (__bf16)b1.x; bfrag[5]  = (__bf16)b1.y;
        bfrag[6]  = (__bf16)b1.z; bfrag[7]  = (__bf16)b1.w;
        bfrag[8]  = (__bf16)b2.x; bfrag[9]  = (__bf16)b2.y;
        bfrag[10] = (__bf16)b2.z; bfrag[11] = (__bf16)b2.w;
        bfrag[12] = (__bf16)b3.x; bfrag[13] = (__bf16)b3.y;
        bfrag[14] = (__bf16)b3.z; bfrag[15] = (__bf16)b3.w;

        acc = __builtin_amdgcn_wmma_f32_16x16x32_bf16(
            false, afrag, false, bfrag, (short)0, acc, false, false);
    }

    // D layout: VGPR r -> M = r + 8*hi, N = lane&15
#pragma unroll
    for (int r = 0; r < 8; ++r) {
        int m = r + hi * 8;
        int o = to * 128 + wid * 16 + m;
        int l = tl * 16 + row;
        float v = acc[r] + bias[o];
        if (act == 1) v = gelu_f(v);
        size_t oi = ((size_t)b * O + o) * L + l;
        if (res) v += res[oi];
        out[oi] = v;
    }
}

// ---------------------------------------------------------------- pooling
// out (B, 4H, L/4): out[b, h*4+s, l'] = x[b, h, l'*4+s]
__global__ void k_rearr_down(const float* __restrict__ x, float* __restrict__ out,
                             int H, int L) {
    size_t i = (size_t)blockIdx.x * blockDim.x + threadIdx.x;
    size_t n = (size_t)BATCH * H * L;
    if (i >= n) return;
    int Lp = L / 4;
    size_t b = i / ((size_t)H * L);
    size_t r = i % ((size_t)H * L);
    size_t ii = r / Lp, l = r % Lp;
    out[i] = x[((size_t)b * H + ii / 4) * L + l * 4 + (ii % 4)];
}

// y (B, Hout*4, Lout/4) -> causal shift by 1 then '(h s) l -> h (l s)'
__global__ void k_up_shift_rearr(const float* __restrict__ y, float* __restrict__ out,
                                 int Hout, int Lout) {
    size_t i = (size_t)blockIdx.x * blockDim.x + threadIdx.x;
    size_t n = (size_t)BATCH * Hout * Lout;
    if (i >= n) return;
    int Lp = Lout / 4;
    size_t b = i / ((size_t)Hout * Lout);
    size_t r = i % ((size_t)Hout * Lout);
    size_t h = r / Lout, l = r % Lout;
    size_t lp = l / 4, ss = l % 4;
    out[i] = (lp == 0) ? 0.f
           : y[((size_t)b * (Hout * 4) + h * 4 + ss) * Lp + lp - 1];
}

// ---------------------------------------------------------------- final LN
__global__ void k_final_ln(const float* __restrict__ u, float* __restrict__ out,
                           const float* __restrict__ g, const float* __restrict__ b,
                           int L) {
    const int H = 128;
    int i = blockIdx.x * blockDim.x + threadIdx.x;
    if (i >= BATCH * L) return;
    int bb = i / L, l = i % L;
    size_t base = (size_t)bb * H * L + l;
    float m = 0.f;
    for (int h = 0; h < H; ++h) m += u[base + (size_t)h * L];
    m *= (1.0f / H);
    float v = 0.f;
    for (int h = 0; h < H; ++h) { float d = u[base + (size_t)h * L] - m; v += d * d; }
    v *= (1.0f / H);
    float rs = rsqrtf(v + 1e-5f);
    size_t ob = ((size_t)bb * L + l) * H;
    for (int h = 0; h < H; ++h)
        out[ob + h] = (u[base + (size_t)h * L] - m) * rs * g[h] + b[h];
}

// ================================================================ host side
struct S4P { const float *ln_g,*ln_b,*log_dt,*A_re,*A_im,*B_re,*B_im,*C_re,*C_im,*D,*Wo,*bo; };
struct FFP { const float *ln_g,*ln_b,*W1,*b1,*W2,*b2; };
struct Block { bool is_s4; S4P s4; FFP ff; };
struct UpStage { const float *W, *b; Block blocks[8]; };

static void gemm(const float* W, const float* X, const float* bias, const float* res,
                 float* out, int O, int Kd, int L, int act, hipStream_t s) {
    dim3 g((O / 128) * (L / 16), BATCH);
    k_gemm_wmma<<<g, 256, 0, s>>>(W, X, bias, res, out, O, Kd, L, act);
}

static void run_block(const Block& blk, float* u, int H, int L,
                      float* bufZ, float* bufC, float* bufW,
                      float* Kbuf, float* CtBuf, hipStream_t s) {
    size_t n = (size_t)BATCH * H * L;
    const int T = 256;
    if (blk.is_s4) {
        const S4P& p = blk.s4;
        k_ln_chan<<<(BATCH * L + T - 1) / T, T, 0, s>>>(u, bufZ, p.ln_g, p.ln_b, H, L);
        k_ct_gen<<<(H * NSTATE + T - 1) / T, T, 0, s>>>(p.log_dt, p.A_re, p.A_im,
                                                        p.B_re, p.B_im, p.C_re, p.C_im,
                                                        CtBuf, H);
        k_kgen<<<(unsigned)(((size_t)H * L + T - 1) / T), T, 0, s>>>(CtBuf, Kbuf, H, L);
        k_conv<<<dim3(L / 256, BATCH * H), 256, 0, s>>>(bufZ, Kbuf, bufC, H, L);
        k_dskip_gelu<<<(unsigned)((n + T - 1) / T), T, 0, s>>>(bufC, bufZ, p.D, H, L, n);
        gemm(p.Wo, bufC, p.bo, nullptr, bufW, 2 * H, H, L, 0, s);
        k_glu<<<(unsigned)((n + T - 1) / T), T, 0, s>>>(u, u, bufW, H, L, n);
    } else {
        const FFP& p = blk.ff;
        k_ln_chan<<<(BATCH * L + T - 1) / T, T, 0, s>>>(u, bufZ, p.ln_g, p.ln_b, H, L);
        gemm(p.W1, bufZ, p.b1, nullptr, bufW, 2 * H, H, L, 1, s);
        gemm(p.W2, bufW, p.b2, u, u, H, 2 * H, L, 0, s);
    }
}

extern "C" void kernel_launch(void* const* d_in, const int* in_sizes, int n_in,
                              void* d_out, int out_size, void* d_ws, size_t ws_size,
                              hipStream_t stream) {
    (void)in_sizes; (void)n_in; (void)out_size; (void)ws_size;
    int idx = 0;
    auto nxt = [&]() { return (const float*)d_in[idx++]; };

    // ---- walk params in setup_inputs() dict insertion order
    const float* Xin = nxt();
    const float *dnW[2], *dnB[2];
    for (int i = 0; i < 2; ++i) { dnW[i] = nxt(); dnB[i] = nxt(); }
    auto parseBlock = [&](Block& blk, bool is_s4) {
        blk.is_s4 = is_s4;
        if (is_s4) {
            S4P& p = blk.s4;
            p.ln_g = nxt(); p.ln_b = nxt(); p.log_dt = nxt();
            p.A_re = nxt(); p.A_im = nxt(); p.B_re = nxt(); p.B_im = nxt();
            p.C_re = nxt(); p.C_im = nxt(); p.D = nxt(); p.Wo = nxt(); p.bo = nxt();
        } else {
            FFP& p = blk.ff;
            p.ln_g = nxt(); p.ln_b = nxt();
            p.W1 = nxt(); p.b1 = nxt(); p.W2 = nxt(); p.b2 = nxt();
        }
    };
    Block cblk[8];
    for (int i = 0; i < 8; ++i) parseBlock(cblk[i], (i % 2) == 0);
    UpStage up[2];
    for (int i = 0; i < 2; ++i) {
        up[i].W = nxt(); up[i].b = nxt();
        for (int j = 0; j < 8; ++j) parseBlock(up[i].blocks[j], (j % 2) == 0);
    }
    const float* ng = nxt();
    const float* nb = nxt();

    // ---- workspace carve (floats)
    float* ws = (float*)d_ws;
    const size_t TSZ = 4194304;         // 8*128*4096
    float* uA    = ws;
    float* uB    = ws + 1 * TSZ;
    float* bufZ  = ws + 2 * TSZ;
    float* bufC  = ws + 3 * TSZ;        // conv out / pool scratch
    float* bufX  = ws + 4 * TSZ;        // residual saves
    float* skip0 = ws + 5 * TSZ;
    float* skip1 = ws + 6 * TSZ;
    float* bufW  = ws + 7 * TSZ;        // 2*TSZ (wide 2H intermediates)
    float* Kbuf  = ws + 9 * TSZ;        // 524288
    float* CtBuf = Kbuf + 524288;       // 65536

    const int T = 256;
    int H = 128, L = 4096;
    size_t n = (size_t)BATCH * H * L;

    k_transpose_in<<<(unsigned)((n + T - 1) / T), T, 0, stream>>>(Xin, uA, L, H);
    k_copy<<<(unsigned)((n + T - 1) / T), T, 0, stream>>>(skip0, uA, n);

    // down 0: 128x4096 -> 256x1024
    k_rearr_down<<<(unsigned)((n + T - 1) / T), T, 0, stream>>>(uA, bufC, H, L);
    gemm(dnW[0], bufC, dnB[0], nullptr, uB, 256, 512, 1024, 0, stream);
    H = 256; L = 1024; n = (size_t)BATCH * H * L; std::swap(uA, uB);
    k_copy<<<(unsigned)((n + T - 1) / T), T, 0, stream>>>(skip1, uA, n);

    // down 1: 256x1024 -> 512x256
    k_rearr_down<<<(unsigned)((n + T - 1) / T), T, 0, stream>>>(uA, bufC, H, L);
    gemm(dnW[1], bufC, dnB[1], nullptr, uB, 512, 1024, 256, 0, stream);
    H = 512; L = 256; n = (size_t)BATCH * H * L; std::swap(uA, uB);

    // center blocks + residual
    k_copy<<<(unsigned)((n + T - 1) / T), T, 0, stream>>>(bufX, uA, n);
    for (int i = 0; i < 8; ++i)
        run_block(cblk[i], uA, H, L, bufZ, bufC, bufW, Kbuf, CtBuf, stream);
    k_add<<<(unsigned)((n + T - 1) / T), T, 0, stream>>>(uA, uA, bufX, n);

    // up 0: 512x256 -> 256x1024
    gemm(up[0].W, uA, up[0].b, nullptr, bufW, 1024, 512, 256, 0, stream);
    H = 256; L = 1024; n = (size_t)BATCH * H * L;
    k_up_shift_rearr<<<(unsigned)((n + T - 1) / T), T, 0, stream>>>(bufW, uB, H, L);
    std::swap(uA, uB);
    k_add<<<(unsigned)((n + T - 1) / T), T, 0, stream>>>(uA, uA, skip1, n);
    k_copy<<<(unsigned)((n + T - 1) / T), T, 0, stream>>>(bufX, uA, n);
    for (int i = 0; i < 8; ++i)
        run_block(up[0].blocks[i], uA, H, L, bufZ, bufC, bufW, Kbuf, CtBuf, stream);
    k_add<<<(unsigned)((n + T - 1) / T), T, 0, stream>>>(uA, uA, bufX, n);

    // up 1: 256x1024 -> 128x4096
    gemm(up[1].W, uA, up[1].b, nullptr, bufW, 512, 256, 1024, 0, stream);
    H = 128; L = 4096; n = (size_t)BATCH * H * L;
    k_up_shift_rearr<<<(unsigned)((n + T - 1) / T), T, 0, stream>>>(bufW, uB, H, L);
    std::swap(uA, uB);
    k_add<<<(unsigned)((n + T - 1) / T), T, 0, stream>>>(uA, uA, skip0, n);
    k_copy<<<(unsigned)((n + T - 1) / T), T, 0, stream>>>(bufX, uA, n);
    for (int i = 0; i < 8; ++i)
        run_block(up[1].blocks[i], uA, H, L, bufZ, bufC, bufW, Kbuf, CtBuf, stream);
    k_add<<<(unsigned)((n + T - 1) / T), T, 0, stream>>>(uA, uA, bufX, n);

    // final transpose + LN over channels -> d_out (B, L, 128) f32
    k_final_ln<<<(BATCH * L + T - 1) / T, T, 0, stream>>>(uA, (float*)d_out, ng, nb, L);
}
